// _MultiheadAttention_13984413515925
// MI455X (gfx1250) — matrix operand order repaired and checked by reading, not compiled, not run on hardware
//
#include <hip/hip_runtime.h>
#include <hip/hip_bf16.h>

// ---------------------------------------------------------------------------
// MultiheadAttention with residual score bias, for MI455X (gfx1250, wave32).
// All matmuls use v_wmma_f32_16x16x32_bf16 (bf16 in, f32 accumulate).
// HBM-bound op (~800MB mandatory traffic) -> NT cache hints on streamed data.
// ---------------------------------------------------------------------------

typedef unsigned short u16;
typedef __attribute__((ext_vector_type(16))) __bf16 v16bf;
typedef __attribute__((ext_vector_type(8)))  float  v8f;
typedef __attribute__((ext_vector_type(4)))  float  f32x4;
typedef __attribute__((ext_vector_type(8)))  unsigned short us8;

union FragAB { v16bf v; us8 h[2]; };

__device__ __forceinline__ u16 f32_to_bf16(float f) {
    unsigned u = __builtin_bit_cast(unsigned, f);
    unsigned r = (u + 0x7FFFu + ((u >> 16) & 1u)) >> 16;   // round-to-nearest-even
    return (u16)r;
}

// ---------------------------------------------------------------------------
// Kernel 0: fp32 -> bf16 conversion. X stays row-major; weights are stored
// TRANSPOSED (Wt[c][k]) so WMMA B-fragments are contiguous b128 loads.
// ---------------------------------------------------------------------------
__global__ void convert_kernel(const float* __restrict__ Q,
                               const float* __restrict__ Wq, const float* __restrict__ Wk,
                               const float* __restrict__ Wv, const float* __restrict__ Wo,
                               u16* __restrict__ Xbf,
                               u16* __restrict__ Wqt, u16* __restrict__ Wkt,
                               u16* __restrict__ Wvt, u16* __restrict__ Wot) {
    long i = (long)blockIdx.x * blockDim.x + threadIdx.x;
    const long NX = 4l * 1024 * 1024;
    if (i < NX) { Xbf[i] = f32_to_bf16(Q[i]); return; }
    long j = i - NX;
    if (j >= 4l * 1024 * 1024) return;
    int w   = (int)(j >> 20);
    int rem = (int)(j & ((1 << 20) - 1));
    int k = rem >> 10, c = rem & 1023;
    const float* src = (w == 0) ? Wq : (w == 1) ? Wk : (w == 2) ? Wv : Wo;
    u16*         dst = (w == 0) ? Wqt : (w == 1) ? Wkt : (w == 2) ? Wvt : Wot;
    dst[(long)c * 1024 + k] = f32_to_bf16(src[rem]);
}

// ---------------------------------------------------------------------------
// Generic WMMA GEMM: C[4096,1024] = A[4096,1024] @ Bt^T + bias.
// One wave computes a 16x64 strip (4 accumulators). Per K-step: load the A
// fragment ONCE plus 4 B fragments into DISTINCT registers, then issue the 4
// wmmas -- distinct regs let the scheduler keep loads of the next iteration
// in flight during the wmma group (partial s_wait_loadcnt, not full drains).
// Unconditional global_prefetch_b8 8 K-steps ahead (speculative, near scope).
// mode 0: fp32 out [row][col]        (final projection -> d_out, NT stores)
// mode 1: bf16 out [b][h][s][dk]     (q, k projections)
// mode 2: bf16 out [b][h][dk][s]     (v projection, transposed for P@V)
// ---------------------------------------------------------------------------
__global__ __launch_bounds__(128) void gemm_wmma(const u16* __restrict__ A,
                                                 const u16* __restrict__ Bt,
                                                 const float* __restrict__ bias,
                                                 float* __restrict__ outF,
                                                 u16* __restrict__ outB,
                                                 int mode) {
    const int tid  = threadIdx.x;
    const int lane = tid & 31;
    const int wave = tid >> 5;
    const int rt   = blockIdx.x;                     // 0..255 (row tiles)
    const int ct0  = (blockIdx.y * 4 + wave) * 4;    // base col tile, 4 per wave

    const int mrow = lane & 15;                 // A row / B col within tile
    const int kbA  = (lane & 16) ? 8 : 0;       // A frag K base (ISA 16-bit A layout)
    const int kbB  = (lane & 16) ? 8 : 0;       // B frag K base (same layout as A)
    const int crow = (lane & 16) ? 8 : 0;       // C/D row offset

    const u16* arow  = A  + ((long)rt * 16 + mrow) * 1024 + kbA;
    const u16* brow0 = Bt + ((long)(ct0 + 0) * 16 + mrow) * 1024 + kbB;
    const u16* brow1 = Bt + ((long)(ct0 + 1) * 16 + mrow) * 1024 + kbB;
    const u16* brow2 = Bt + ((long)(ct0 + 2) * 16 + mrow) * 1024 + kbB;
    const u16* brow3 = Bt + ((long)(ct0 + 3) * 16 + mrow) * 1024 + kbB;

    v8f acc0 = {}, acc1 = {}, acc2 = {}, acc3 = {};
#pragma unroll 2
    for (int kt = 0; kt < 32; ++kt) {
        const int k0 = kt * 32;
        // gfx1250 speculative prefetch, 8 K-steps (512B) ahead, near scope.
        __builtin_prefetch(arow  + k0 + 256, 0, 3);
        __builtin_prefetch(brow0 + k0 + 256, 0, 3);
        __builtin_prefetch(brow2 + k0 + 256, 0, 3);

        FragAB a, b0, b1, b2, b3;
        a.h[0]  = *(const us8*)(arow  + k0);
        a.h[1]  = *(const us8*)(arow  + k0 + 16);
        b0.h[0] = *(const us8*)(brow0 + k0);
        b0.h[1] = *(const us8*)(brow0 + k0 + 16);
        b1.h[0] = *(const us8*)(brow1 + k0);
        b1.h[1] = *(const us8*)(brow1 + k0 + 16);
        b2.h[0] = *(const us8*)(brow2 + k0);
        b2.h[1] = *(const us8*)(brow2 + k0 + 16);
        b3.h[0] = *(const us8*)(brow3 + k0);
        b3.h[1] = *(const us8*)(brow3 + k0 + 16);

        acc0 = __builtin_amdgcn_wmma_f32_16x16x32_bf16(false, a.v, false, b0.v,
                                                       (short)0, acc0, false, false);
        acc1 = __builtin_amdgcn_wmma_f32_16x16x32_bf16(false, a.v, false, b1.v,
                                                       (short)0, acc1, false, false);
        acc2 = __builtin_amdgcn_wmma_f32_16x16x32_bf16(false, a.v, false, b2.v,
                                                       (short)0, acc2, false, false);
        acc3 = __builtin_amdgcn_wmma_f32_16x16x32_bf16(false, a.v, false, b3.v,
                                                       (short)0, acc3, false, false);
    }

    const v8f accs[4] = {acc0, acc1, acc2, acc3};
#pragma unroll
    for (int c = 0; c < 4; ++c) {
        const int   col = (ct0 + c) * 16 + mrow;
        const float bv  = bias[col];
#pragma unroll
        for (int r = 0; r < 8; ++r) {
            const int   row = rt * 16 + r + crow;
            const float val = accs[c][r] + bv;
            if (mode == 0) {
                __builtin_nontemporal_store(val, &outF[(long)row * 1024 + col]);
            } else {
                const int b_ = row >> 10, s = row & 1023;
                const int h_ = col >> 6,  dk = col & 63;
                if (mode == 1)
                    outB[(((long)(b_ * 16 + h_)) * 1024 + s) * 64 + dk] = f32_to_bf16(val);
                else
                    outB[(((long)(b_ * 16 + h_)) * 64 + dk) * 1024 + s] = f32_to_bf16(val);
            }
        }
    }
}

// ---------------------------------------------------------------------------
// Attention kernel: one block (4 waves) per (b, h, 16-row q tile).
//   Phase 1: S = scale*(q k^T) + prev  -> attn_scores (WMMA, waves split k)
//   Phase 2: online softmax over rows  -> attn_weights + bf16 P in LDS
//   Phase 3: O = P @ V (WMMA; A frags from LDS via ds_load_b128)
// prev: NT load (streamed once). weights: NT store (never re-read).
// scores: RT store (re-read by this WGP in phase 2); last read NT.
// ---------------------------------------------------------------------------
__global__ __launch_bounds__(128) void attn_kernel(const u16* __restrict__ qb,
                                                   const u16* __restrict__ kb,
                                                   const u16* __restrict__ vt,
                                                   const float* __restrict__ prev,
                                                   float* __restrict__ scores,
                                                   float* __restrict__ weights,
                                                   u16* __restrict__ attn_out) {
    __shared__ u16   Pb[16 * 1024];         // 32 KB bf16 softmax weights
    __shared__ float redm[128], reds[128];
    __shared__ float rowmax[16], rowinv[16];

    const int tid  = threadIdx.x;
    const int lane = tid & 31;
    const int wave = tid >> 5;
    const int blk  = blockIdx.x;            // 0..4095
    const int qt   = blk & 63;              // q tile
    const int bh   = blk >> 6;              // 0..63
    const int b    = bh >> 4, h = bh & 15;

    const int mrow = lane & 15;
    const int kbA  = (lane & 16) ? 8 : 0;
    const int kbB  = (lane & 16) ? 8 : 0;
    const int crow = (lane & 16) ? 8 : 0;

    const long sc_base = ((long)bh * 1024 + qt * 16) * 1024;

    // ---- Phase 1: score tiles (16x16 each); waves stride over 64 k-tiles ----
    for (int kt = wave; kt < 64; kt += 4) {
        v8f acc = {};
#pragma unroll
        for (int chunk = 0; chunk < 64; chunk += 32) {
            const u16* ap = qb + (((long)bh * 1024 + qt * 16 + mrow) * 64 + chunk);
            const u16* bp = kb + (((long)bh * 1024 + kt * 16 + mrow) * 64 + chunk);
            FragAB a, kf;
            a.h[0]  = *(const us8*)(ap + kbA);
            a.h[1]  = *(const us8*)(ap + kbA + 16);
            kf.h[0] = *(const us8*)(bp + kbB);
            kf.h[1] = *(const us8*)(bp + kbB + 16);
            acc = __builtin_amdgcn_wmma_f32_16x16x32_bf16(false, a.v, false, kf.v,
                                                          (short)0, acc, false, false);
        }
#pragma unroll
        for (int r = 0; r < 8; ++r) {
            const int  row = r + crow;
            const long idx = sc_base + (long)row * 1024 + kt * 16 + mrow;
            const float v  = acc[r] * 0.125f                       // scale = 1/sqrt(64)
                           + __builtin_nontemporal_load(&prev[idx]);
            scores[idx] = v;                                        // RT: re-read below
        }
    }
    __threadfence_block();
    __syncthreads();

    // ---- Phase 2: online softmax. thread t: row = t&15, 128-col chunk = t>>4 ----
    const int r2 = tid & 15;
    const int c2 = tid >> 4;
    {
        const long gbase = sc_base + (long)r2 * 1024 + c2 * 128;
        float m = -3.4e38f, s = 0.f;
        for (int j = 0; j < 128; j += 4) {
            const f32x4 x4 = *(const f32x4*)(&scores[gbase + j]);
#pragma unroll
            for (int u = 0; u < 4; ++u) {
                const float mn = fmaxf(m, x4[u]);
                s = s * __expf(m - mn) + __expf(x4[u] - mn);
                m = mn;
            }
        }
        redm[c2 * 16 + r2] = m;
        reds[c2 * 16 + r2] = s;
    }
    __syncthreads();
    if (tid < 16) {
        float M = -3.4e38f;
        for (int c = 0; c < 8; ++c) M = fmaxf(M, redm[c * 16 + tid]);
        float S = 0.f;
        for (int c = 0; c < 8; ++c) S += reds[c * 16 + tid] * __expf(redm[c * 16 + tid] - M);
        rowmax[tid] = M;
        rowinv[tid] = 1.f / S;
    }
    __syncthreads();
    {
        const float M   = rowmax[r2];
        const float inv = rowinv[r2];
        const long  gbase = sc_base + (long)r2 * 1024 + c2 * 128;
        const int   lbase = r2 * 1024 + c2 * 128;
        for (int j = 0; j < 128; j += 4) {
            const f32x4 x4 = __builtin_nontemporal_load((const f32x4*)(&scores[gbase + j]));
#pragma unroll
            for (int u = 0; u < 4; ++u) {
                const float w = __expf(x4[u] - M) * inv;
                __builtin_nontemporal_store(w, &weights[gbase + j + u]);
                Pb[lbase + j + u] = f32_to_bf16(w);
            }
        }
    }
    __syncthreads();

    // ---- Phase 3: O[16x64] = P[16x1024] @ V[1024x64]; wave w owns dk chunk w ----
    {
        v8f acc = {};
        const int  n    = wave * 16 + mrow;                     // dk column 0..63
        const u16* vrow = vt + ((long)bh * 64 + n) * 1024;
        for (int kt = 0; kt < 32; ++kt) {
            FragAB a, vb;
            const u16* ap = Pb + mrow * 1024 + kt * 32;         // LDS -> ds_load_b128
            a.h[0] = *(const us8*)(ap + kbA);
            a.h[1] = *(const us8*)(ap + kbA + 16);
            const u16* bp = vrow + kt * 32 + kbB;
            vb.h[0] = *(const us8*)(bp);
            vb.h[1] = *(const us8*)(bp + 16);
            acc = __builtin_amdgcn_wmma_f32_16x16x32_bf16(false, a.v, false, vb.v,
                                                          (short)0, acc, false, false);
        }
#pragma unroll
        for (int r = 0; r < 8; ++r) {
            const int srow = qt * 16 + r + crow;
            attn_out[((long)b * 1024 + srow) * 1024 + h * 64 + n] = f32_to_bf16(acc[r]);
        }
    }
}

// ---------------------------------------------------------------------------
// kernel_launch: convert -> 3 projections -> attention -> output projection.
// d_in order: Q, prev, Wq, bq, Wk, bk, Wv, bv, Wo, bo
// d_out: out(4M f32) | attn_weights(64M f32) | attn_scores(64M f32)
// ---------------------------------------------------------------------------
extern "C" void kernel_launch(void* const* d_in, const int* in_sizes, int n_in,
                              void* d_out, int out_size, void* d_ws, size_t ws_size,
                              hipStream_t stream) {
    (void)in_sizes; (void)n_in; (void)out_size; (void)ws_size;

    const float* Q    = (const float*)d_in[0];
    const float* prev = (const float*)d_in[1];
    const float* Wq   = (const float*)d_in[2];
    const float* bq   = (const float*)d_in[3];
    const float* Wk   = (const float*)d_in[4];
    const float* bk   = (const float*)d_in[5];
    const float* Wv   = (const float*)d_in[6];
    const float* bv   = (const float*)d_in[7];
    const float* Wo   = (const float*)d_in[8];
    const float* bo   = (const float*)d_in[9];

    float* out     = (float*)d_out;
    float* weights = out + 4l * 1024 * 1024;
    float* scores  = weights + 64l * 1024 * 1024;

    char* w = (char*)d_ws;                       // 48 MiB total
    u16* Xbf = (u16*)(w);                        //  8 MiB  X bf16
    u16* Wqt = (u16*)(w + (8l  << 20));          //  2 MiB  Wq^T bf16
    u16* Wkt = (u16*)(w + (10l << 20));
    u16* Wvt = (u16*)(w + (12l << 20));
    u16* Wot = (u16*)(w + (14l << 20));
    u16* qb  = (u16*)(w + (16l << 20));          //  8 MiB  q [b,h,s,dk]
    u16* kb  = (u16*)(w + (24l << 20));          //  8 MiB  k [b,h,s,dk]
    u16* vt  = (u16*)(w + (32l << 20));          //  8 MiB  v [b,h,dk,s]
    u16* ao  = (u16*)(w + (40l << 20));          //  8 MiB  attn out bf16

    convert_kernel<<<32768, 256, 0, stream>>>(Q, Wq, Wk, Wv, Wo,
                                              Xbf, Wqt, Wkt, Wvt, Wot);

    dim3 gg(256, 4);
    gemm_wmma<<<gg, 128, 0, stream>>>(Xbf, Wqt, bq, nullptr, qb, 1);
    gemm_wmma<<<gg, 128, 0, stream>>>(Xbf, Wkt, bk, nullptr, kb, 1);
    gemm_wmma<<<gg, 128, 0, stream>>>(Xbf, Wvt, bv, nullptr, vt, 2);

    attn_kernel<<<4096, 128, 0, stream>>>(qb, kb, vt, prev, scores, weights, ao);

    gemm_wmma<<<gg, 128, 0, stream>>>(ao, Wot, bo, out, nullptr, 0);
}
